// CAttentionLegacy_71038759075976
// MI455X (gfx1250) — matrix operand order repaired and checked by reading, not compiled, not run on hardware
//
#include <hip/hip_runtime.h>

// ---------------------------------------------------------------------------
// CDNA5 (gfx1250) channel-attention block, bf16 WMMA implicit-GEMM pipeline.
// Dominant cost: dense 3x3 conv (576->576) ~783 GFLOP -> v_wmma_f32_16x16x32_bf16
// with workgroup-shared async LDS staging of activations (ASYNCcnt path).
// ---------------------------------------------------------------------------

typedef __attribute__((ext_vector_type(16))) __bf16 v16bf;
typedef __attribute__((ext_vector_type(8)))  __bf16 v8bf;
typedef __attribute__((ext_vector_type(8)))  float  v8f;
typedef __attribute__((ext_vector_type(4)))  int    v4i;

#define DEV __device__ __forceinline__

constexpr int BATCH  = 8;
constexpr int CDIM   = 192;          // dim
constexpr int C3     = 576;          // 3*dim
constexpr int K9     = 5184;         // 9*576  (conv3x3 GEMM K)
constexpr int HH     = 128;
constexpr int WW     = 128;
constexpr int P      = HH * WW;      // 16384 pixels per image
constexpr int NHEADS = 6;
constexpr int NSTEP  = 162;          // K9 / 32

#if defined(__gfx1250__) && __has_builtin(__builtin_amdgcn_global_load_async_to_lds_b128)
#define HAVE_ASYNC 1
#else
#define HAVE_ASYNC 0
#endif

#if HAVE_ASYNC
// Param types per hipcc diagnostic: (v4i AS1*, v4i AS3*, imm offset, imm cpol)
#define ASYNC_LD16(g, l)                                                      \
  __builtin_amdgcn_global_load_async_to_lds_b128(                             \
      (__attribute__((address_space(1))) v4i*)(g),                            \
      (__attribute__((address_space(3))) v4i*)(l), 0, 0)
#define WAIT_ASYNC2() asm volatile("s_wait_asynccnt 0x2" ::: "memory")
#define WAIT_ASYNC0() asm volatile("s_wait_asynccnt 0x0" ::: "memory")
#else
#define WAIT_ASYNC2()
#define WAIT_ASYNC0()
#endif

union VB { v16bf v; v8bf h[2]; unsigned u[8]; };

DEV unsigned short f2bf(float f) {                 // fp32 -> bf16 RNE
  union { float f; unsigned u; } a; a.f = f;
  unsigned r = a.u + 0x7fffu + ((a.u >> 16) & 1u);
  return (unsigned short)(r >> 16);
}
DEV float bf2f(unsigned short h) {
  union { unsigned u; float f; } a; a.u = ((unsigned)h) << 16;
  return a.f;
}

// A operand (16xK tile, row-major [row][ld]): lane<16 -> row=lane,
// K = k0..k0+15 contiguous; lane>=16 -> K = k0+16..k0+31.
DEV v16bf load_a16(const unsigned short* base, int ld, int k0, int lane) {
  const unsigned short* p = base + (size_t)(lane & 15) * ld + k0 + ((lane >> 4) << 4);
  VB u;
  u.h[0] = *(const v8bf*)p;
  u.h[1] = *(const v8bf*)(p + 8);
  return u.v;
}
// B operand (Kx16 tile, stored as [n][K] row-major): lane<16 -> col=lane,
// K = k0..k0+15 contiguous; lane>=16 -> K = k0+16..k0+31.
DEV v16bf load_b16(const unsigned short* base, int ld, int k0, int lane) {
  const unsigned short* p = base + (size_t)(lane & 15) * ld + k0 + ((lane >> 4) << 4);
  VB u;
  u.h[0] = *(const v8bf*)p;
  u.h[1] = *(const v8bf*)(p + 8);
  return u.v;
}
DEV v8f zero8() { v8f z; for (int i = 0; i < 8; i++) z[i] = 0.f; return z; }
DEV v8f wmma_bf16(v16bf a, v16bf b, v8f c) {
  return __builtin_amdgcn_wmma_f32_16x16x32_bf16(false, a, false, b, (short)0, c,
                                                 false, false);
}

// ---------------------------------------------------------------------------
// K1: convert weights to bf16; reshape W_dw OIHW[o][c][ky][kx] -> [o][r*576+c]
// ---------------------------------------------------------------------------
__global__ void k_convert_weights(const float* wqkv, const float* wdw, const float* wproj,
                                  unsigned short* bq, unsigned short* bd, unsigned short* bp) {
  const int n1 = C3 * CDIM, n2 = C3 * K9, n3 = CDIM * CDIM;
  for (int i = blockIdx.x * blockDim.x + threadIdx.x; i < n1 + n2 + n3;
       i += gridDim.x * blockDim.x) {
    if (i < n1) {
      bq[i] = f2bf(wqkv[i]);
    } else if (i < n1 + n2) {
      int j = i - n1;
      int o = j / K9, k = j % K9;
      int r = k / C3, c = k % C3;
      bd[j] = f2bf(wdw[((size_t)o * C3 + c) * 9 + r]);
    } else {
      int j = i - n1 - n2;
      bp[j] = f2bf(wproj[j]);
    }
  }
}

// ---------------------------------------------------------------------------
// K2: x NCHW fp32 -> pixel-major bf16  xT[b][p][c]   (LDS-tiled transpose)
// ---------------------------------------------------------------------------
__global__ void __launch_bounds__(256) k_x_transpose(const float* x, unsigned short* xT) {
  __shared__ unsigned short tile[32][CDIM + 4];
  int blk = blockIdx.x;                 // b*512 + ptile
  int b = blk >> 9, p0 = (blk & 511) << 5;
  int t = threadIdx.x;
  for (int e = t; e < CDIM * 32; e += 256) {
    int c = e >> 5, pl = e & 31;        // pl fastest -> coalesced fp32 reads
    tile[pl][c] = f2bf(x[((size_t)b * CDIM + c) * P + p0 + pl]);
  }
  __syncthreads();
  for (int e = t; e < 32 * CDIM; e += 256) {
    int pl = e / CDIM, c = e % CDIM;    // c fastest -> coalesced bf16 writes
    xT[((size_t)b * P + p0 + pl) * CDIM + c] = tile[pl][c];
  }
}

// ---------------------------------------------------------------------------
// K3: qkv1 = W_qkv(576x192) @ x(192xP), stored transposed q1T[b][p][576].
//     Register double-buffered so WMMAs overlap in-flight loads.
// ---------------------------------------------------------------------------
__global__ void __launch_bounds__(256) k_gemm_qkv1(const unsigned short* bq,
                                                   const unsigned short* xT,
                                                   unsigned short* q1T) {
  int wave = blockIdx.x * 8 + (threadIdx.x >> 5);
  int lane = threadIdx.x & 31;
  int b = wave / (36 * 256), rem = wave % (36 * 256);
  int m0 = (rem >> 8) << 4, n0 = (rem & 255) << 6;
  const unsigned short* abase = bq + (size_t)m0 * CDIM;
  const unsigned short* bbase = xT + ((size_t)b * P + n0) * CDIM;

  v8f acc[4]; for (int t = 0; t < 4; t++) acc[t] = zero8();
  v16bf a = load_a16(abase, CDIM, 0, lane);
  v16bf bv[4];
#pragma unroll
  for (int t = 0; t < 4; t++) bv[t] = load_b16(bbase + (size_t)(t * 16) * CDIM, CDIM, 0, lane);

  for (int k0 = 0; k0 < CDIM; k0 += 32) {
    int kn = (k0 + 32 < CDIM) ? k0 + 32 : 0;       // wrap: always-valid prefetch
    v16bf an = load_a16(abase, CDIM, kn, lane);
    v16bf bn[4];
#pragma unroll
    for (int t = 0; t < 4; t++) bn[t] = load_b16(bbase + (size_t)(t * 16) * CDIM, CDIM, kn, lane);
#pragma unroll
    for (int t = 0; t < 4; t++) acc[t] = wmma_bf16(a, bv[t], acc[t]);
    a = an;
#pragma unroll
    for (int t = 0; t < 4; t++) bv[t] = bn[t];
  }
  int ln = lane & 15, hi = lane >> 4;
  for (int t = 0; t < 4; t++)
    for (int i = 0; i < 8; i++) {
      int m = m0 + i + (hi << 3);
      int n = n0 + (t << 4) + ln;
      q1T[((size_t)b * P + n) * C3 + m] = f2bf(acc[t][i]);   // transposed store
    }
}

// ---------------------------------------------------------------------------
// K4: dense 3x3 conv as implicit GEMM, K = 5184 (the 783-GFLOP stage).
//     Block = 4 waves = 4 M-tiles sharing one 64-pixel strip.
//     B tile (64px x 32K, tap-shifted, zero-padded) staged to LDS with
//     GLOBAL_LOAD_ASYNC_TO_LDS_B128, double-buffered against s_wait_asynccnt.
// ---------------------------------------------------------------------------
DEV void stage_step(const unsigned short* q1b, int n0, int s, int t,
                    unsigned short (*sbuf)[40], unsigned short* sdummy) {
  int r = s / 18, kk = s - r * 18;
  int dy = r / 3 - 1, dx = r % 3 - 1;
  int pl = t >> 1;                        // local pixel 0..63
  int half = (t & 1) << 4;                // element offset 0 or 16
  int y = (n0 >> 7) + dy;
  int x = (n0 & 127) + pl + dx;
  bool ok = ((unsigned)y < (unsigned)HH) && ((unsigned)x < (unsigned)WW);
  unsigned short* lp = &sbuf[pl][half];
#if HAVE_ASYNC
  if (ok) {
    const unsigned short* gp = q1b + ((size_t)((y << 7) + x)) * C3 + kk * 32 + half;
    ASYNC_LD16(gp, lp);
    ASYNC_LD16(gp + 8, lp + 8);
  } else {
    ASYNC_LD16(q1b, sdummy);              // keep ASYNCcnt invariant (2/thread)
    ASYNC_LD16(q1b, sdummy);
    VB z; for (int q = 0; q < 8; q++) z.u[q] = 0u;
    *(v8bf*)lp = z.h[0];
    *(v8bf*)(lp + 8) = z.h[0];
  }
#else
  (void)sdummy;
  VB d;
  if (ok) {
    const unsigned short* gp = q1b + ((size_t)((y << 7) + x)) * C3 + kk * 32 + half;
    d.h[0] = *(const v8bf*)gp;
    d.h[1] = *(const v8bf*)(gp + 8);
  } else {
    for (int q = 0; q < 8; q++) d.u[q] = 0u;
  }
  *(v8bf*)lp = d.h[0];
  *(v8bf*)(lp + 8) = d.h[1];
#endif
}

__global__ void __launch_bounds__(128) k_gemm_conv3(const unsigned short* bd,
                                                    const unsigned short* q1T,
                                                    unsigned short* qk_cm,
                                                    unsigned short* vT) {
  __shared__ unsigned short sB[2][64][40];   // padded rows: 80B stride, no 16-row conflicts
  __shared__ unsigned short sdummy[8];

  int t = threadIdx.x;
  int wv = t >> 5, lane = t & 31;
  int blk = blockIdx.x;                      // b*2304 + mg*256 + ns
  int b = blk / 2304, rem = blk % 2304;
  int mg = rem >> 8, ns = rem & 255;
  int m0 = (mg * 4 + wv) << 4;               // this wave's M-tile
  int n0 = ns << 6;                          // shared 64-pixel strip
  const unsigned short* abase = bd + (size_t)m0 * K9;
  const unsigned short* q1b = q1T + (size_t)b * P * C3;
  int ln = lane & 15, hi = lane >> 4;

  v8f acc[4]; for (int tt = 0; tt < 4; tt++) acc[tt] = zero8();

  stage_step(q1b, n0, 0, t, sB[0], sdummy);
  v16bf a = load_a16(abase, K9, 0, lane);

  for (int s = 0; s < NSTEP; s++) {
    int buf = s & 1;
    if (s + 1 < NSTEP) stage_step(q1b, n0, s + 1, t, sB[1 - buf], sdummy);
    int kn = (s + 1 < NSTEP) ? (s + 1) * 32 : 0;
    __builtin_prefetch(abase + (size_t)ln * K9 + kn, 0, 0);   // global_prefetch_b8
    v16bf an = load_a16(abase, K9, kn, lane);
    if (s + 1 < NSTEP) { WAIT_ASYNC2(); } else { WAIT_ASYNC0(); }
    __syncthreads();                       // buf[s&1] fully staged for all waves
#pragma unroll
    for (int tt = 0; tt < 4; tt++) {
      const unsigned short* lp = &sB[buf][(tt << 4) + ln][hi << 4];
      VB ub;
      ub.h[0] = *(const v8bf*)lp;          // ds_load_b128 x2
      ub.h[1] = *(const v8bf*)(lp + 8);
      acc[tt] = wmma_bf16(a, ub.v, acc[tt]);
    }
    __syncthreads();                       // readers done before next overwrite
    a = an;
  }

  if (m0 < 384) {          // q,k -> channel-major [b][o][p]
    for (int tt = 0; tt < 4; tt++)
      for (int i = 0; i < 8; i++) {
        int m = m0 + i + (hi << 3), n = n0 + (tt << 4) + ln;
        qk_cm[((size_t)b * 384 + m) * P + n] = f2bf(acc[tt][i]);
      }
  } else {                 // v -> pixel-major [b][p][192]
    for (int tt = 0; tt < 4; tt++)
      for (int i = 0; i < 8; i++) {
        int m = m0 + i + (hi << 3) - 384, n = n0 + (tt << 4) + ln;
        vT[((size_t)b * P + n) * CDIM + m] = f2bf(acc[tt][i]);
      }
  }
}

// ---------------------------------------------------------------------------
// K5: per (b,head): fused unnormalized q.k^T (32x32, K=16384) + L2 norms +
//     temperature + row softmax -> bf16 attn[co][cd] (WMMA B operand layout).
// ---------------------------------------------------------------------------
__global__ void __launch_bounds__(256) k_attn(const unsigned short* qk_cm,
                                              const float* temperature,
                                              unsigned short* attnb) {
  __shared__ unsigned short sq[32][256];
  __shared__ unsigned short sk[32][256];
  __shared__ float s_at[32][33];
  __shared__ float s_qn[32], s_kn[32];
  int bh = blockIdx.x;
  int b = bh / NHEADS, h = bh % NHEADS;
  int t = threadIdx.x;
  const unsigned short* qb = qk_cm + ((size_t)b * 384 + h * 32) * P;
  const unsigned short* kb = qk_cm + ((size_t)b * 384 + 192 + h * 32) * P;

  int i0 = (t * 4) >> 5;
  int j0 = (t * 4) & 31;
  float a0 = 0, a1 = 0, a2 = 0, a3 = 0, qs = 0, ks = 0;

  for (int c0 = 0; c0 < P; c0 += 256) {
    __syncthreads();
    for (int s = 0; s < 32; s++) {
      sq[s][t] = qb[(size_t)s * P + c0 + t];
      sk[s][t] = kb[(size_t)s * P + c0 + t];
    }
    __syncthreads();
    for (int c = 0; c < 256; c++) {
      float qv = bf2f(sq[i0][c]);
      a0 += qv * bf2f(sk[j0][c]);
      a1 += qv * bf2f(sk[j0 + 1][c]);
      a2 += qv * bf2f(sk[j0 + 2][c]);
      a3 += qv * bf2f(sk[j0 + 3][c]);
    }
    if (t < 32) {
      for (int c = 0; c < 256; c++) {
        float qq = bf2f(sq[t][c]); qs += qq * qq;
        float kk = bf2f(sk[t][c]); ks += kk * kk;
      }
    }
  }
  s_at[i0][j0] = a0; s_at[i0][j0 + 1] = a1;
  s_at[i0][j0 + 2] = a2; s_at[i0][j0 + 3] = a3;
  if (t < 32) { s_qn[t] = qs; s_kn[t] = ks; }
  __syncthreads();

  if (t < 32) {
    float temp = temperature[h];
    float qinv = 1.f / fmaxf(sqrtf(s_qn[t]), 1e-12f);
    float logits[32];
    float mx = -1e30f;
    for (int j = 0; j < 32; j++) {
      float kinv = 1.f / fmaxf(sqrtf(s_kn[j]), 1e-12f);
      float l = s_at[t][j] * qinv * kinv * temp;
      logits[j] = l;
      mx = fmaxf(mx, l);
    }
    float sum = 0.f;
    for (int j = 0; j < 32; j++) { float e = __expf(logits[j] - mx); logits[j] = e; sum += e; }
    float inv = 1.f / sum;
    for (int j = 0; j < 32; j++)
      attnb[(size_t)bh * 1024 + t * 32 + j] = f2bf(logits[j] * inv);
  }
}

// ---------------------------------------------------------------------------
// K6: out^T[p][head*32+co] = v^T[p][cd] @ attn[co][cd]  -- K=32 = one WMMA.
// ---------------------------------------------------------------------------
__global__ void __launch_bounds__(256) k_gemm_av(const unsigned short* vT,
                                                 const unsigned short* attnb,
                                                 unsigned short* oT) {
  int wave = blockIdx.x * 8 + (threadIdx.x >> 5);
  int lane = threadIdx.x & 31;
  int b = wave / (NHEADS * 1024), rem = wave % (NHEADS * 1024);
  int h = rem >> 10, p0 = (rem & 1023) << 4;
  int ln = lane & 15, hi = lane >> 4;

  const unsigned short* ab = vT + ((size_t)b * P + p0) * CDIM + h * 32;
  const unsigned short* bb = attnb + (size_t)(b * NHEADS + h) * 1024;

  v16bf a  = load_a16(ab, CDIM, 0, lane);
  v16bf b0 = load_b16(bb, 32, 0, lane);
  v16bf b1 = load_b16(bb + 16 * 32, 32, 0, lane);
  v8f c0 = wmma_bf16(a, b0, zero8());
  v8f c1 = wmma_bf16(a, b1, zero8());

  for (int i = 0; i < 8; i++) {
    int p = p0 + i + (hi << 3);
    size_t row = ((size_t)b * P + p) * CDIM + h * 32;
    oT[row + ln]      = f2bf(c0[i]);
    oT[row + 16 + ln] = f2bf(c1[i]);
  }
}

// ---------------------------------------------------------------------------
// K7: y = W_proj(192x192) @ out(192xP), fp32 NCHW to d_out. Double-buffered.
// ---------------------------------------------------------------------------
__global__ void __launch_bounds__(256) k_gemm_proj(const unsigned short* bp,
                                                   const unsigned short* oT,
                                                   float* out) {
  int wave = blockIdx.x * 8 + (threadIdx.x >> 5);
  int lane = threadIdx.x & 31;
  int b = wave / (12 * 256), rem = wave % (12 * 256);
  int m0 = (rem >> 8) << 4, n0 = (rem & 255) << 6;
  const unsigned short* abase = bp + (size_t)m0 * CDIM;
  const unsigned short* bbase = oT + ((size_t)b * P + n0) * CDIM;

  v8f acc[4]; for (int t = 0; t < 4; t++) acc[t] = zero8();
  v16bf a = load_a16(abase, CDIM, 0, lane);
  v16bf bv[4];
#pragma unroll
  for (int t = 0; t < 4; t++) bv[t] = load_b16(bbase + (size_t)(t * 16) * CDIM, CDIM, 0, lane);

  for (int k0 = 0; k0 < CDIM; k0 += 32) {
    int kn = (k0 + 32 < CDIM) ? k0 + 32 : 0;
    v16bf an = load_a16(abase, CDIM, kn, lane);
    v16bf bn[4];
#pragma unroll
    for (int t = 0; t < 4; t++) bn[t] = load_b16(bbase + (size_t)(t * 16) * CDIM, CDIM, kn, lane);
#pragma unroll
    for (int t = 0; t < 4; t++) acc[t] = wmma_bf16(a, bv[t], acc[t]);
    a = an;
#pragma unroll
    for (int t = 0; t < 4; t++) bv[t] = bn[t];
  }
  int ln = lane & 15, hi = lane >> 4;
  for (int t = 0; t < 4; t++)
    for (int i = 0; i < 8; i++) {
      int m = m0 + i + (hi << 3), n = n0 + (t << 4) + ln;
      out[((size_t)b * CDIM + m) * P + n] = acc[t][i];
    }
}

// ---------------------------------------------------------------------------
extern "C" void kernel_launch(void* const* d_in, const int* in_sizes, int n_in,
                              void* d_out, int out_size, void* d_ws, size_t ws_size,
                              hipStream_t stream) {
  const float* x     = (const float*)d_in[0];
  const float* wqkv  = (const float*)d_in[1];
  const float* wdw   = (const float*)d_in[2];
  const float* wproj = (const float*)d_in[3];
  const float* temp  = (const float*)d_in[4];
  float* out = (float*)d_out;

  char* ws = (char*)d_ws;
  size_t off = 0;
  auto alloc = [&](size_t bytes) {
    size_t o = off;
    off += (bytes + 255) & ~(size_t)255;
    return o;
  };
  unsigned short* bq   = (unsigned short*)(ws + alloc((size_t)C3 * CDIM * 2));
  unsigned short* bd   = (unsigned short*)(ws + alloc((size_t)C3 * K9 * 2));
  unsigned short* bp   = (unsigned short*)(ws + alloc((size_t)CDIM * CDIM * 2));
  unsigned short* xT   = (unsigned short*)(ws + alloc((size_t)BATCH * P * CDIM * 2));
  unsigned short* q1T  = (unsigned short*)(ws + alloc((size_t)BATCH * P * C3 * 2));
  unsigned short* qkcm = (unsigned short*)(ws + alloc((size_t)BATCH * 384 * P * 2));
  unsigned short* vT   = (unsigned short*)(ws + alloc((size_t)BATCH * P * CDIM * 2));
  unsigned short* atb  = (unsigned short*)(ws + alloc((size_t)BATCH * NHEADS * 1024 * 2));
  unsigned short* oT   = (unsigned short*)(ws + alloc((size_t)BATCH * P * CDIM * 2));

  hipLaunchKernelGGL(k_convert_weights, dim3(4096), dim3(256), 0, stream,
                     wqkv, wdw, wproj, bq, bd, bp);
  hipLaunchKernelGGL(k_x_transpose, dim3(BATCH * 512), dim3(256), 0, stream, x, xT);
  hipLaunchKernelGGL(k_gemm_qkv1, dim3(BATCH * 36 * 256 / 8), dim3(256), 0, stream,
                     bq, xT, q1T);
  hipLaunchKernelGGL(k_gemm_conv3, dim3(BATCH * 9 * 256), dim3(128), 0, stream,
                     bd, q1T, qkcm, vT);
  hipLaunchKernelGGL(k_attn, dim3(BATCH * NHEADS), dim3(256), 0, stream,
                     qkcm, temp, atb);
  hipLaunchKernelGGL(k_gemm_av, dim3(BATCH * NHEADS * 1024 / 8), dim3(256), 0, stream,
                     vT, atb, oT);
  hipLaunchKernelGGL(k_gemm_proj, dim3(BATCH * 12 * 256 / 8), dim3(256), 0, stream,
                     bp, oT, out);
}
